// AMP_Net_16363825397986
// MI455X (gfx1250) — matrix operand order repaired and checked by reading, not compiled, not run on hardware
//
#include <hip/hip_runtime.h>

#define NN 512
#define AA 30
#define DD 729
#define LL 3
#define BB 2
#define PI_F 3.14159265358979f

typedef _Float16 f16;
typedef __attribute__((ext_vector_type(16))) _Float16 v16h;
typedef __attribute__((ext_vector_type(8)))  _Float16 v8h;
typedef __attribute__((ext_vector_type(8)))  float    v8f;

// ---------------------------------------------------------------- trig table
__global__ void k_trig(const float* __restrict__ theta, float* cs, float* sn) {
    int t = threadIdx.x;
    if (t < AA) {
        float th = theta[t];
        cs[t] = cosf(th);
        sn[t] = sinf(th);
    }
}

// ------------------------------------------------- forward Radon projection
// out[b,a,s] = sum_t bilinear(img_b, row= s'*sn + t'*cs + c, col= s'*cs - t'*sn + c)
__global__ void k_radon(const float* __restrict__ img,
                        const float* __restrict__ cs, const float* __restrict__ sn,
                        float* __restrict__ out) {
    int gid = blockIdx.x * blockDim.x + threadIdx.x;
    if (gid >= BB * AA * DD) return;
    int s = gid % DD;
    int a = (gid / DD) % AA;
    int b = gid / (AA * DD);
    const float* im = img + (size_t)b * NN * NN;
    const float c  = (NN - 1) * 0.5f;
    const float sp = (float)s - (DD - 1) * 0.5f;
    float cv = cs[a], sv = sn[a];
    float row0 = sp * sv + c;
    float col0 = sp * cv + c;
    float acc = 0.0f;
    for (int t = 0; t < DD; ++t) {
        float tp  = (float)t - (DD - 1) * 0.5f;
        float row = row0 + tp * cv;
        float col = col0 - tp * sv;
        int y0 = (int)floorf(row);
        int x0 = (int)floorf(col);
        float wy = row - (float)y0;
        float wx = col - (float)x0;
        int y1 = y0 + 1, x1 = x0 + 1;
        float v00 = (y0 >= 0 && y0 < NN && x0 >= 0 && x0 < NN) ? im[y0 * NN + x0] : 0.0f;
        float v01 = (y0 >= 0 && y0 < NN && x1 >= 0 && x1 < NN) ? im[y0 * NN + x1] : 0.0f;
        float v10 = (y1 >= 0 && y1 < NN && x0 >= 0 && x0 < NN) ? im[y1 * NN + x0] : 0.0f;
        float v11 = (y1 >= 0 && y1 < NN && x1 >= 0 && x1 < NN) ? im[y1 * NN + x1] : 0.0f;
        acc += (1.0f - wy) * ((1.0f - wx) * v00 + wx * v01)
             +         wy  * ((1.0f - wx) * v10 + wx * v11);
    }
    out[gid] = acc;
}

// ------------------------------------------------------- ramp filter (spatial)
// FFT ramp filter == circular conv with g: g[0]=0.5, g[d odd]=-2/(pi d)^2, else 0.
// out[s] = base[s] - (pi/2A) * (g * p)[s]
__global__ void k_filter(const float* __restrict__ p,
                         const float* __restrict__ base,
                         float* __restrict__ out) {
    int gid = blockIdx.x * blockDim.x + threadIdx.x;
    if (gid >= BB * AA * DD) return;
    int s = gid % DD;
    const float* pr = p + (gid - s);
    float acc = (PI_F / (4.0f * AA)) * pr[s];
    float sum = 0.0f;
    for (int d = 1; d < DD; d += 2) {
        float v = 0.0f;
        int lo = s - d, hi = s + d;
        if (lo >= 0) v += pr[lo];
        if (hi < DD) v += pr[hi];
        sum += v / (float)(d * d);
    }
    acc += (-1.0f / (AA * PI_F)) * sum;
    out[gid] = base[gid] - acc;
}

// ---------------------------------------------- conv1: 1 -> 32 ch, NHWC f16 out
__global__ void k_conv1(const float* __restrict__ X,
                        const float* __restrict__ w1, const float* __restrict__ b1,
                        f16* __restrict__ h) {
    __shared__ float wsm[32 * 9];
    __shared__ float bsm[32];
    int tid = threadIdx.x;
    for (int i = tid; i < 288; i += blockDim.x) wsm[i] = w1[i];
    if (tid < 32) bsm[tid] = b1[tid];
    __syncthreads();
    int gid = blockIdx.x * blockDim.x + tid;
    int x = gid % NN;
    int y = (gid / NN) % NN;
    int b = gid / (NN * NN);
    const float* im = X + (size_t)b * NN * NN;
    float v[9];
#pragma unroll
    for (int ky = 0; ky < 3; ++ky)
#pragma unroll
        for (int kx = 0; kx < 3; ++kx) {
            int py = y + ky - 1, px = x + kx - 1;
            v[ky * 3 + kx] = (py >= 0 && py < NN && px >= 0 && px < NN) ? im[py * NN + px] : 0.0f;
        }
    f16* op = h + ((size_t)b * NN * NN + (size_t)y * NN + x) * 32;
    for (int o = 0; o < 32; ++o) {
        float acc = bsm[o];
#pragma unroll
        for (int k = 0; k < 9; ++k) acc += wsm[o * 9 + k] * v[k];
        acc = acc > 0.0f ? acc : 0.0f;
        op[o] = (f16)acc;
    }
}

// -------------------------- repack 32x32x3x3 OIHW weights into WMMA A layout
// wp[tap][mtile][lane][h] lane l: row l%16, K = 16*(l/16) + h (contiguous k order, same as B)
__global__ void k_packw(const float* __restrict__ w, f16* __restrict__ wp) {
    int t = blockIdx.x * blockDim.x + threadIdx.x;
    if (t >= 9216) return;
    int h     = t & 15;
    int lane  = (t >> 4) & 31;
    int mtile = (t >> 9) & 1;
    int tap   = t >> 10;
    int M, K;
    if (lane < 16) { M = lane;      K = h;      }  // K in {0-15}
    else           { M = lane - 16; K = h + 16; }  // K in {16-31}
    int o = mtile * 16 + M;
    wp[t] = (f16)w[(o * 32 + K) * 9 + tap];
}

// --------------------------------------- WMMA conv: 32 -> 32 channels, f16 NHWC
// Each wave: 16 pixels x 32 out-channels via 2 accumulators, 9 taps x 2 WMMA.
__global__ void __launch_bounds__(256) k_conv_wmma(const f16* __restrict__ in,
                                                   const f16* __restrict__ wp,
                                                   const float* __restrict__ bias,
                                                   f16* __restrict__ out) {
    int lane = threadIdx.x & 31;
    int wave = threadIdx.x >> 5;
    int bx   = blockIdx.x;          // BB * 512 rows * 4 blocks/row
    int b    = bx >> 11;            // 2048 blocks per batch image
    int rem  = bx & 2047;
    int y    = rem >> 2;
    int x0   = (rem & 3) * 128 + wave * 16;
    const f16* inb = in + (size_t)b * NN * NN * 32;
    int pxl = x0 + (lane & 15);     // this lane's pixel column
    int c0  = (lane >> 4) * 16;     // K half: lanes 0-15 -> ch 0-15, 16-31 -> ch 16-31
    v8f acc0 = {}; v8f acc1 = {};
#pragma unroll
    for (int ky = 0; ky < 3; ++ky) {
        int py = y + ky - 1;
        bool yok = (py >= 0) && (py < NN);
#pragma unroll
        for (int kx = 0; kx < 3; ++kx) {
            int tap = ky * 3 + kx;
            int px  = pxl + kx - 1;
            bool ok = yok && (px >= 0) && (px < NN);
            v16h bmat;
            if (ok) {
                bmat = *(const v16h*)(inb + ((size_t)py * NN + px) * 32 + c0);
            } else {
                v16h z = {}; bmat = z;
            }
            v16h a0 = *(const v16h*)(wp + (tap * 2 + 0) * 512 + lane * 16);
            v16h a1 = *(const v16h*)(wp + (tap * 2 + 1) * 512 + lane * 16);
            acc0 = __builtin_amdgcn_wmma_f32_16x16x32_f16(false, a0, false, bmat,
                                                          (short)0, acc0, false, false);
            acc1 = __builtin_amdgcn_wmma_f32_16x16x32_f16(false, a1, false, bmat,
                                                          (short)0, acc1, false, false);
        }
    }
    // D layout: VGPR r, lane<16 -> (ch r, pix lane); lane>=16 -> (ch r+8, pix lane-16)
    int cb  = (lane >= 16) ? 8 : 0;
    f16* op = out + ((size_t)b * NN * NN + (size_t)y * NN + pxl) * 32;
    v8h o0, o1;
#pragma unroll
    for (int r = 0; r < 8; ++r) {
        float v0 = acc0[r] + bias[cb + r];       v0 = v0 > 0.0f ? v0 : 0.0f;
        float v1 = acc1[r] + bias[16 + cb + r];  v1 = v1 > 0.0f ? v1 : 0.0f;
        o0[r] = (f16)v0;
        o1[r] = (f16)v1;
    }
    *(v8h*)(op + cb)      = o0;
    *(v8h*)(op + 16 + cb) = o1;
}

// ---------------------------------------------- conv4: 32 -> 1 ch, no bias
__global__ void k_conv4(const f16* __restrict__ h, const float* __restrict__ w4,
                        float* __restrict__ noise) {
    __shared__ float wsm[288];
    int tid = threadIdx.x;
    for (int i = tid; i < 288; i += blockDim.x) wsm[i] = w4[i];
    __syncthreads();
    int gid = blockIdx.x * blockDim.x + tid;
    int x = gid % NN;
    int y = (gid / NN) % NN;
    int b = gid / (NN * NN);
    const f16* hb = h + (size_t)b * NN * NN * 32;
    float acc = 0.0f;
#pragma unroll
    for (int ky = 0; ky < 3; ++ky)
#pragma unroll
        for (int kx = 0; kx < 3; ++kx) {
            int py = y + ky - 1, px = x + kx - 1;
            if (py >= 0 && py < NN && px >= 0 && px < NN) {
                const f16* pp = hb + ((size_t)py * NN + px) * 32;
                int tap = ky * 3 + kx;
                for (int i = 0; i < 32; ++i) acc += (float)pp[i] * wsm[i * 9 + tap];
            }
        }
    noise[gid] = acc;
}

// ------------------- fused update: X += noise + step * backproject(q)
__global__ void k_update(float* __restrict__ X, const float* __restrict__ noise,
                         const float* __restrict__ q,
                         const float* __restrict__ cs, const float* __restrict__ sn,
                         const float* __restrict__ step_p) {
    int gid = blockIdx.x * blockDim.x + threadIdx.x;
    int x = gid % NN;
    int y = (gid / NN) % NN;
    int b = gid / (NN * NN);
    float xs = (float)x - (NN - 1) * 0.5f;
    float ys = (float)y - (NN - 1) * 0.5f;
    const float* qb = q + (size_t)b * AA * DD;
    float acc = 0.0f;
    for (int a = 0; a < AA; ++a) {
        float idx = cs[a] * xs + sn[a] * ys + (DD - 1) * 0.5f;
        int lo = (int)floorf(idx);
        float w = idx - (float)lo;
        float v = 0.0f;
        if (lo >= 0 && lo < DD)           v += qb[a * DD + lo] * (1.0f - w);
        if (lo + 1 >= 0 && lo + 1 < DD)   v += qb[a * DD + lo + 1] * w;
        acc += v;
    }
    float st = *step_p;
    X[gid] = X[gid] + noise[gid] + st * acc;
}

// --------------------------------- output: tuple of 3 identical copies of X
__global__ void k_out(const float* __restrict__ X, float* __restrict__ out) {
    int gid = blockIdx.x * blockDim.x + threadIdx.x;
    if (gid >= BB * NN * NN) return;
    float v = X[gid];
    out[gid]                    = v;
    out[gid + BB * NN * NN]     = v;
    out[gid + 2 * BB * NN * NN] = v;
}

extern "C" void kernel_launch(void* const* d_in, const int* in_sizes, int n_in,
                              void* d_out, int out_size, void* d_ws, size_t ws_size,
                              hipStream_t stream) {
    (void)in_sizes; (void)n_in; (void)out_size; (void)ws_size;
    const float* x0    = (const float*)d_in[1];
    const float* sino  = (const float*)d_in[2];
    const float* theta = (const float*)d_in[3];
    const float* w1    = (const float*)d_in[5];
    const float* b1    = (const float*)d_in[6];
    const float* w2    = (const float*)d_in[7];
    const float* b2    = (const float*)d_in[8];
    const float* w3    = (const float*)d_in[9];
    const float* b3    = (const float*)d_in[10];
    const float* w4    = (const float*)d_in[11];
    const float* steps = (const float*)d_in[12];

    char* wsb = (char*)d_ws;
    size_t ofs = 0;
    auto alloc = [&](size_t bytes) -> char* {
        char* p = wsb + ofs;
        ofs += (bytes + 255) & ~(size_t)255;
        return p;
    };
    float* X     = (float*)alloc((size_t)BB * NN * NN * 4);
    float* noise = (float*)alloc((size_t)BB * NN * NN * 4);
    float* sp    = (float*)alloc((size_t)BB * AA * DD * 4);
    float* r     = (float*)alloc((size_t)BB * AA * DD * 4);
    float* csb   = (float*)alloc(AA * 4);
    float* snb   = (float*)alloc(AA * 4);
    f16*   h1    = (f16*)alloc((size_t)BB * NN * NN * 32 * 2);
    f16*   h2    = (f16*)alloc((size_t)BB * NN * NN * 32 * 2);
    f16*   wp    = (f16*)alloc(9216 * 2);

    hipMemcpyAsync(X, x0, (size_t)BB * NN * NN * 4, hipMemcpyDeviceToDevice, stream);
    k_trig<<<1, 32, 0, stream>>>(theta, csb, snb);

    const int npx  = BB * NN * NN;
    const int nsin = BB * AA * DD;
    const int sino_blocks = (nsin + 255) / 256;
    dim3 blk(256);

    for (int n = 0; n < LL; ++n) {
        // sino-domain residual: r = sino - ramp(radon(X))
        k_radon <<<sino_blocks, blk, 0, stream>>>(X, csb, snb, sp);
        k_filter<<<sino_blocks, blk, 0, stream>>>(sp, sino, r);
        // denoiser
        k_conv1<<<npx / 256, blk, 0, stream>>>(X, w1 + n * 288, b1 + n * 32, h1);
        k_packw<<<(9216 + 255) / 256, blk, 0, stream>>>(w2 + n * 9216, wp);
        k_conv_wmma<<<BB * NN * 4, blk, 0, stream>>>(h1, wp, b2 + n * 32, h2);
        k_packw<<<(9216 + 255) / 256, blk, 0, stream>>>(w3 + n * 9216, wp);
        k_conv_wmma<<<BB * NN * 4, blk, 0, stream>>>(h2, wp, b3 + n * 32, h1);
        k_conv4<<<npx / 256, blk, 0, stream>>>(h1, w4 + n * 288, noise);
        // q = r - ramp(radon(noise))   (linearity fuses both backprojections)
        k_radon <<<sino_blocks, blk, 0, stream>>>(noise, csb, snb, sp);
        k_filter<<<sino_blocks, blk, 0, stream>>>(sp, r, r);
        // X += noise + step * backproject(q)
        k_update<<<npx / 256, blk, 0, stream>>>(X, noise, r, csb, snb, steps + n);
    }
    k_out<<<npx / 256, blk, 0, stream>>>(X, (float*)d_out);
}
